// MultiCabecaAtencao_63763084476445
// MI455X (gfx1250) — matrix operand order repaired and checked by reading, not compiled, not run on hardware
//
#include <hip/hip_runtime.h>
#include <hip/hip_bf16.h>

// ---------------------------------------------------------------------------
// Types for CDNA5 WMMA (wave32): v_wmma_f32_16x16x32_bf16
// ---------------------------------------------------------------------------
typedef __attribute__((ext_vector_type(16))) __bf16 v16bf;
typedef __attribute__((ext_vector_type(8)))  __bf16 v8bf;
typedef __attribute__((ext_vector_type(8)))  float  v8f;

union Frag16 { v16bf v; v8bf h[2]; };

#define BM 128
#define BN 128
#define BK 32
#define LDSK 40   // padded row stride (elements); keeps 16B alignment (80B rows)

// 16-byte async copy global -> LDS (no VGPR staging, tracked by ASYNCcnt)
__device__ __forceinline__ void async_copy16(void* lds_dst, const void* g_src) {
#if __has_builtin(__builtin_amdgcn_global_load_async_to_lds_b128)
  typedef __attribute__((ext_vector_type(4))) int i4;
  typedef __attribute__((address_space(1))) i4 gi4;
  typedef __attribute__((address_space(3))) i4 li4;
  __builtin_amdgcn_global_load_async_to_lds_b128(
      (gi4*)(unsigned long long)g_src,
      (li4*)(unsigned)(unsigned long long)lds_dst, 0, 0);
#else
  asm volatile("global_load_async_to_lds_b128 %0, %1, off"
               :
               : "v"((unsigned)(unsigned long long)lds_dst),
                 "v"((unsigned long long)g_src)
               : "memory");
#endif
}

__device__ __forceinline__ void wait_async0() {
#if __has_builtin(__builtin_amdgcn_s_wait_asynccnt)
  __builtin_amdgcn_s_wait_asynccnt(0);
#else
  asm volatile("s_wait_asynccnt 0" ::: "memory");
#endif
}

// ---------------------------------------------------------------------------
// C[M,N] = scale * (A[M,K] @ Bt[N,K]^T) + bias
// A row-major [M][K] bf16, Bt row-major [N][K] bf16 (i.e. B transposed).
// Double-buffered LDS with async global->LDS copies.
// ---------------------------------------------------------------------------
template <int OUT_BF16, int HAS_BIAS>
__global__ __launch_bounds__(256)
void gemm_bt_wmma(const __bf16* __restrict__ A, const __bf16* __restrict__ Bt,
                  void* __restrict__ C, const float* __restrict__ bias,
                  int Mdim, int Ndim, int Kdim, float scale,
                  long sA, long sB, long sC) {
  __shared__ __bf16 Als[2][BM][LDSK];
  __shared__ __bf16 Bls[2][BN][LDSK];

  const int tid  = threadIdx.x;
  const int lane = tid & 31;
  const int wave = tid >> 5;
  const int waveM = (wave & 3) * 32;   // 4 waves along M
  const int waveN = (wave >> 2) * 64;  // 2 waves along N

  A  += (long)blockIdx.z * sA;
  Bt += (long)blockIdx.z * sB;
  const long cbase = (long)blockIdx.z * sC;

  const int m0 = blockIdx.y * BM;
  const int n0 = blockIdx.x * BN;

  const __bf16* gA = A  + (long)m0 * Kdim;
  const __bf16* gB = Bt + (long)n0 * Kdim;

  // Per-thread chunk coordinates: 512 x 16B chunks per matrix, 2 per thread.
  const int r0c = (tid + 0)   >> 2;           // rows 0..127
  const int r1c = (tid + 256) >> 2;
  const int c0c = ((tid + 0)   & 3) << 3;     // 0,8,16,24
  const int c1c = ((tid + 256) & 3) << 3;

  auto issue_tile = [&](int buf, int k0) {
    async_copy16(&Als[buf][r0c][c0c], gA + (long)r0c * Kdim + k0 + c0c);
    async_copy16(&Bls[buf][r0c][c0c], gB + (long)r0c * Kdim + k0 + c0c);
    async_copy16(&Als[buf][r1c][c1c], gA + (long)r1c * Kdim + k0 + c1c);
    async_copy16(&Bls[buf][r1c][c1c], gB + (long)r1c * Kdim + k0 + c1c);
  };

  v8f acc[2][4] = {};

  // Prologue: fill buffer 0
  issue_tile(0, 0);
  wait_async0();
  __syncthreads();

  int buf = 0;
  for (int k0 = 0; k0 < Kdim; k0 += BK) {
    const int nk = k0 + BK;
    // Kick off next tile into the other buffer while we compute this one.
    if (nk < Kdim) issue_tile(buf ^ 1, nk);
    if (nk + BK < Kdim) {  // warm L2 two tiles ahead (global_prefetch_b8)
      __builtin_prefetch(gA + (long)r0c * Kdim + nk + BK + c0c, 0, 1);
      __builtin_prefetch(gB + (long)r0c * Kdim + nk + BK + c0c, 0, 1);
    }

    // A fragments: lane L holds row (L%16); K chunks at kb and kb+16, kb=8*(L/16)
    v16bf af[2];
#pragma unroll
    for (int i = 0; i < 2; ++i) {
      int row = waveM + i * 16 + (lane & 15);
      int kb  = (lane >> 4) << 3;
      Frag16 f;
      f.h[0] = *(const v8bf*)(&Als[buf][row][kb]);
      f.h[1] = *(const v8bf*)(&Als[buf][row][kb + 16]);
      af[i] = f.v;
    }
    // B fragments: lane L holds column (L%16); K chunks at kb and kb+16, kb=8*(L/16)
    v16bf bfm[4];
#pragma unroll
    for (int j = 0; j < 4; ++j) {
      int col = waveN + j * 16 + (lane & 15);
      int kb  = (lane >> 4) << 3;
      Frag16 f;
      f.h[0] = *(const v8bf*)(&Bls[buf][col][kb]);
      f.h[1] = *(const v8bf*)(&Bls[buf][col][kb + 16]);
      bfm[j] = f.v;
    }

#pragma unroll
    for (int i = 0; i < 2; ++i)
#pragma unroll
      for (int j = 0; j < 4; ++j)
        acc[i][j] = __builtin_amdgcn_wmma_f32_16x16x32_bf16(
            false, af[i], false, bfm[j], (short)0, acc[i][j], false, false);

    // Handoff: my async writes for next tile done, then block-wide barrier.
    if (nk < Kdim) wait_async0();
    __syncthreads();
    buf ^= 1;
  }

  // Epilogue. C/D layout: VGPR r, lane L -> (M = r + 8*(L/16), N = L%16)
#pragma unroll
  for (int i = 0; i < 2; ++i) {
#pragma unroll
    for (int j = 0; j < 4; ++j) {
      int rowBase = m0 + waveM + i * 16 + ((lane >> 4) << 3);
      int col     = n0 + waveN + j * 16 + (lane & 15);
      float bv = HAS_BIAS ? bias[col] : 0.0f;
#pragma unroll
      for (int r = 0; r < 8; ++r) {
        float val = acc[i][j][r] * scale + bv;
        long  o   = cbase + (long)(rowBase + r) * Ndim + col;
        if (OUT_BF16) ((__bf16*)C)[o] = (__bf16)val;
        else          ((float*)C)[o]  = val;
      }
    }
  }
}

// ---------------------------------------------------------------------------
// float -> bf16 elementwise convert (grid-stride)
// ---------------------------------------------------------------------------
__global__ __launch_bounds__(256)
void cvt_f32_bf16(const float* __restrict__ in, __bf16* __restrict__ out, long n) {
  for (long i = (long)blockIdx.x * 256 + threadIdx.x; i < n;
       i += (long)gridDim.x * 256)
    out[i] = (__bf16)in[i];
}

// ---------------------------------------------------------------------------
// Transpose f32 [R][C] -> bf16 [C][R]   (for weight matrices)
// ---------------------------------------------------------------------------
__global__ __launch_bounds__(256)
void transpose_w_bf16(const float* __restrict__ in, __bf16* __restrict__ out,
                      int R, int Ccols) {
  __shared__ float tile[32][33];
  int r0 = blockIdx.y * 32, c0 = blockIdx.x * 32;
  int tx = threadIdx.x & 31, ty = threadIdx.x >> 5;
#pragma unroll
  for (int i = 0; i < 32; i += 8)
    tile[ty + i][tx] = in[(long)(r0 + ty + i) * Ccols + c0 + tx];
  __syncthreads();
#pragma unroll
  for (int i = 0; i < 32; i += 8)
    out[(long)(c0 + ty + i) * R + r0 + tx] = (__bf16)tile[tx][ty + i];
}

// ---------------------------------------------------------------------------
// Transpose bf16 [R][C] -> bf16 [C][R], batched (for V)
// ---------------------------------------------------------------------------
__global__ __launch_bounds__(256)
void transpose_v_bf16(const __bf16* __restrict__ in, __bf16* __restrict__ out,
                      int R, int Ccols, long sIn, long sOut) {
  __shared__ __bf16 tile[32][40];
  in  += (long)blockIdx.z * sIn;
  out += (long)blockIdx.z * sOut;
  int r0 = blockIdx.y * 32, c0 = blockIdx.x * 32;
  int tx = threadIdx.x & 31, ty = threadIdx.x >> 5;
#pragma unroll
  for (int i = 0; i < 32; i += 8)
    tile[ty + i][tx] = in[(long)(r0 + ty + i) * Ccols + c0 + tx];
  __syncthreads();
#pragma unroll
  for (int i = 0; i < 32; i += 8)
    out[(long)(c0 + ty + i) * R + r0 + tx] = tile[tx][ty + i];
}

// ---------------------------------------------------------------------------
// Row softmax: f32 scores [rows][ncols] -> bf16 probs, one block per row.
// ---------------------------------------------------------------------------
__global__ __launch_bounds__(256)
void softmax_bf16(const float* __restrict__ S, __bf16* __restrict__ P, int ncols) {
  long row = blockIdx.x;
  const float* s = S + row * (long)ncols;
  __bf16*      p = P + row * (long)ncols;
  int tid = threadIdx.x;
  __shared__ float red[8];

  float m = -3.4e38f;
  for (int c = tid; c < ncols; c += 256) m = fmaxf(m, s[c]);
#pragma unroll
  for (int off = 16; off > 0; off >>= 1) m = fmaxf(m, __shfl_xor(m, off, 32));
  if ((tid & 31) == 0) red[tid >> 5] = m;
  __syncthreads();
  if (tid < 32) {
    float v = (tid < 8) ? red[tid] : -3.4e38f;
#pragma unroll
    for (int off = 4; off > 0; off >>= 1) v = fmaxf(v, __shfl_xor(v, off, 32));
    if (tid == 0) red[0] = v;
  }
  __syncthreads();
  float rowmax = red[0];
  __syncthreads();

  float sum = 0.0f;
  for (int c = tid; c < ncols; c += 256) sum += __expf(s[c] - rowmax);
#pragma unroll
  for (int off = 16; off > 0; off >>= 1) sum += __shfl_xor(sum, off, 32);
  if ((tid & 31) == 0) red[tid >> 5] = sum;
  __syncthreads();
  if (tid < 32) {
    float v = (tid < 8) ? red[tid] : 0.0f;
#pragma unroll
    for (int off = 4; off > 0; off >>= 1) v += __shfl_xor(v, off, 32);
    if (tid == 0) red[0] = v;
  }
  __syncthreads();
  float inv = 1.0f / red[0];
  for (int c = tid; c < ncols; c += 256)
    p[c] = (__bf16)(__expf(s[c] - rowmax) * inv);
}

// ---------------------------------------------------------------------------
// Orchestration
// ---------------------------------------------------------------------------
extern "C" void kernel_launch(void* const* d_in, const int* in_sizes, int n_in,
                              void* d_out, int out_size, void* d_ws, size_t ws_size,
                              hipStream_t stream) {
  const int B = 4, S = 2048, D = 1024;
  const int M = B * S;  // 8192 rows

  const float* x  = (const float*)d_in[0];
  const float* Wq = (const float*)d_in[1];
  const float* bq = (const float*)d_in[2];
  const float* Wk = (const float*)d_in[3];
  const float* bk = (const float*)d_in[4];
  const float* Wv = (const float*)d_in[5];
  const float* bv = (const float*)d_in[6];
  const float* Wo = (const float*)d_in[7];
  const float* bo = (const float*)d_in[8];
  float* out = (float*)d_out;

  char* ws = (char*)d_ws;
  size_t off = 0;
  auto carve = [&](size_t bytes) {
    char* p = ws + off;
    off += (bytes + 255) & ~(size_t)255;
    return (void*)p;
  };
  __bf16* xb  = (__bf16*)carve((size_t)M * D * 2);
  __bf16* Wqt = (__bf16*)carve((size_t)D * D * 2);
  __bf16* Wkt = (__bf16*)carve((size_t)D * D * 2);
  __bf16* Wvt = (__bf16*)carve((size_t)D * D * 2);
  __bf16* Wot = (__bf16*)carve((size_t)D * D * 2);
  __bf16* qb  = (__bf16*)carve((size_t)M * D * 2);
  __bf16* kb  = (__bf16*)carve((size_t)M * D * 2);
  __bf16* vb  = (__bf16*)carve((size_t)M * D * 2);
  __bf16* vt  = (__bf16*)carve((size_t)M * D * 2);
  float*  sc  = (float*)carve((size_t)B * S * S * 4);
  __bf16* at  = (__bf16*)carve((size_t)B * S * S * 2);
  __bf16* cx  = (__bf16*)carve((size_t)M * D * 2);

  // 1) x -> bf16 ; weights -> bf16 transposed ([N][K])
  cvt_f32_bf16<<<dim3(4096), 256, 0, stream>>>(x, xb, (long)M * D);
  dim3 gw(D / 32, D / 32, 1);
  transpose_w_bf16<<<gw, 256, 0, stream>>>(Wq, Wqt, D, D);
  transpose_w_bf16<<<gw, 256, 0, stream>>>(Wk, Wkt, D, D);
  transpose_w_bf16<<<gw, 256, 0, stream>>>(Wv, Wvt, D, D);
  transpose_w_bf16<<<gw, 256, 0, stream>>>(Wo, Wot, D, D);

  // 2) Projections: q/k/v = xb @ W^T + b  (bf16 out)
  dim3 gp(D / BN, M / BM, 1);
  gemm_bt_wmma<1, 1><<<gp, 256, 0, stream>>>(xb, Wqt, qb, bq, M, D, D, 1.0f, 0, 0, 0);
  gemm_bt_wmma<1, 1><<<gp, 256, 0, stream>>>(xb, Wkt, kb, bk, M, D, D, 1.0f, 0, 0, 0);
  gemm_bt_wmma<1, 1><<<gp, 256, 0, stream>>>(xb, Wvt, vb, bv, M, D, D, 1.0f, 0, 0, 0);

  // 3) scores = q @ k^T / sqrt(64)  (f32 out, batched over B)
  dim3 gs(S / BN, S / BM, B);
  gemm_bt_wmma<0, 0><<<gs, 256, 0, stream>>>(qb, kb, sc, nullptr, S, S, D, 0.125f,
                                             (long)S * D, (long)S * D, (long)S * S);

  // 4) softmax rows -> bf16 probs
  softmax_bf16<<<dim3(B * S), 256, 0, stream>>>(sc, at, S);

  // 5) v^T per batch, then ctx = attn @ v  (bf16 out, batched)
  transpose_v_bf16<<<dim3(D / 32, S / 32, B), 256, 0, stream>>>(
      vb, vt, S, D, (long)S * D, (long)S * D);
  dim3 gc(D / BN, S / BM, B);
  gemm_bt_wmma<1, 0><<<gc, 256, 0, stream>>>(at, vt, cx, nullptr, S, D, S, 1.0f,
                                             (long)S * S, (long)S * D, (long)S * D);

  // 6) out = ctx @ Wo^T + bo  (f32 out)
  dim3 go(D / BN, M / BM, 1);
  gemm_bt_wmma<0, 1><<<go, 256, 0, stream>>>(cx, Wot, out, bo, M, D, D, 1.0f, 0, 0, 0);
}
